// CommNetModule_83013127897258
// MI455X (gfx1250) — matrix operand order repaired and checked by reading, not compiled, not run on hardware
//
#include <hip/hip_runtime.h>

// CommNet forward for MI455X (gfx1250): bf16 WMMA, wave-per-batch-element.
// B=16384, N=32 agents, OBS=64, H=64, A=16, R=2 comm rounds.

#define NB    16384
#define NA    32
#define OBSD  64
#define HD    64
#define AD    16
#define RR    2
#define WAVES 8
#define SSTR  136   // staging row stride (bf16 elems); 272B rows, 16B aligned

typedef __bf16 bf16_t;
typedef __attribute__((ext_vector_type(16))) __bf16 v16bf;
typedef __attribute__((ext_vector_type(8)))  __bf16 v8bf;
typedef __attribute__((ext_vector_type(8)))  float  v8f;

// Native bf16 convert: backend emits v_cvt_*bf16_f32 (1 VALU op vs 4 for
// the software RNE sequence).
__device__ __forceinline__ bf16_t f2bf(float f) { return (bf16_t)f; }

__device__ __forceinline__ unsigned pack2(float a, float b) {
    unsigned short lo = __builtin_bit_cast(unsigned short, (bf16_t)a);
    unsigned short hi = __builtin_bit_cast(unsigned short, (bf16_t)b);
    return (unsigned)lo | ((unsigned)hi << 16);
}

// Assemble a 16-element bf16 fragment from two 16-byte LDS chunks.
__device__ __forceinline__ v16bf load_frag(const bf16_t* p0, const bf16_t* p1) {
    v8bf lo = *(const v8bf*)p0;
    v8bf hi = *(const v8bf*)p1;
    v16bf r;
#pragma unroll
    for (int i = 0; i < 8; ++i) { r[i] = lo[i]; r[i + 8] = hi[i]; }
    return r;
}

__global__ __launch_bounds__(32 * WAVES) void commnet_fwd(
    const float* __restrict__ obs,    const float* __restrict__ enc_w,
    const float* __restrict__ enc_b,  const float* __restrict__ comm_w,
    const float* __restrict__ comm_b, const float* __restrict__ out_w1,
    const float* __restrict__ out_b1, const float* __restrict__ out_w2,
    const float* __restrict__ out_b2, const int*   __restrict__ avail,
    float* __restrict__ q_out)
{
    // Transposed bf16 weights (row = output col, contiguous K) + f32 biases.
    // comm weights use the interleaved-K permutation: k' = 2f (h), 2f+1 (msg).
    __shared__ bf16_t s_encT[HD * OBSD];            //  8 KB
    __shared__ bf16_t s_commT[RR][HD * 2 * HD];     // 32 KB
    __shared__ bf16_t s_w1T[HD * HD];               //  8 KB
    __shared__ bf16_t s_w2T[AD * HD];               //  2 KB
    __shared__ float  s_encb[HD], s_commb[RR][HD], s_b1[HD], s_b2[AD];
    __shared__ bf16_t s_stage[WAVES][NA * SSTR];    // ~68 KB activation staging

    const int tid = threadIdx.x;

    // ---- cooperative weight staging: f32 global -> transposed bf16 LDS ----
    for (int i = tid; i < OBSD * HD; i += 32 * WAVES) {
        int o = i / HD, h = i % HD;
        s_encT[h * OBSD + o] = f2bf(enc_w[i]);
    }
#pragma unroll
    for (int r = 0; r < RR; ++r)
        for (int i = tid; i < 2 * HD * HD; i += 32 * WAVES) {
            int k = i / HD, h = i % HD;
            int kp = (k < HD) ? (2 * k) : (2 * (k - HD) + 1);  // interleave h/msg
            s_commT[r][h * (2 * HD) + kp] = f2bf(comm_w[r * 2 * HD * HD + i]);
        }
    for (int i = tid; i < HD * HD; i += 32 * WAVES) {
        int k = i / HD, h = i % HD;
        s_w1T[h * HD + k] = f2bf(out_w1[i]);
    }
    for (int i = tid; i < HD * AD; i += 32 * WAVES) {
        int h = i / AD, a = i % AD;
        s_w2T[a * HD + h] = f2bf(out_w2[i]);
    }
    for (int i = tid; i < HD; i += 32 * WAVES) {
        s_encb[i]     = enc_b[i];
        s_b1[i]       = out_b1[i];
        s_commb[0][i] = comm_b[i];
        s_commb[1][i] = comm_b[HD + i];
    }
    for (int i = tid; i < AD; i += 32 * WAVES) s_b2[i] = out_b2[i];
    __syncthreads();

    const int  wave = tid >> 5;
    const int  lane = tid & 31;
    const int  half = lane >> 4;     // K/row-half selector per WMMA layout
    const int  lr   = lane & 15;
    const long bidx = (long)blockIdx.x * WAVES + wave;   // one batch elem / wave
    bf16_t* stage   = s_stage[wave];

    const int* avb = avail + bidx * (NA * AD);
    __builtin_prefetch(avb, 0, 1);   // global_prefetch_b8: mask needed at epilogue

    // ---- stage obs rows (32 x 64 f32 -> bf16), coalesced float4 loads,
    //      packed ds_store_b64 writes ----
    const float* orow = obs + bidx * (NA * OBSD);
#pragma unroll
    for (int i = 0; i < 16; ++i) {
        int idx = i * 32 + lane;     // float4 index within 32x16 grid
        int row = idx >> 4;
        int c   = (idx & 15) * 4;
        const float4 v = *(const float4*)(orow + row * OBSD + c);
        uint2 p;
        p.x = pack2(v.x, v.y);
        p.y = pack2(v.z, v.w);
        *(uint2*)(stage + row * SSTR + c) = p;     // 8B aligned
    }

    v8f acc[2][4];   // 32x64 f32 activations in C/D layout (2 row x 4 col tiles)

    // ---- encoder GEMM: h = relu(obs @ enc_w + enc_b), K=64 ----
#pragma unroll
    for (int mt = 0; mt < 2; ++mt)
#pragma unroll
        for (int nt = 0; nt < 4; ++nt) {
            v8f c{};
#pragma unroll
            for (int kt = 0; kt < 2; ++kt) {
                const bf16_t* ap = stage + (mt * 16 + lr) * SSTR + kt * 32 + half * 16;
                v16bf a = load_frag(ap, ap + 8);
                const bf16_t* bp = s_encT + (nt * 16 + lr) * OBSD + kt * 32 + half * 16;
                v16bf b = load_frag(bp, bp + 8);
                c = __builtin_amdgcn_wmma_f32_16x16x32_bf16(
                        false, a, false, b, (short)0, c, false, false);
            }
            float bias = s_encb[nt * 16 + lr];
#pragma unroll
            for (int v = 0; v < 8; ++v) c[v] = fmaxf(c[v] + bias, 0.0f);
            acc[mt][nt] = c;
        }

    // ---- comm rounds: msg = (sum_n h - h)/31 ; h = relu([h|msg] @ W_r + b_r)
    //      combined K is interleaved (h,msg per feature) -> packed b32 stores ----
#pragma unroll
    for (int r = 0; r < RR; ++r) {
        float csum[4];
#pragma unroll
        for (int nt = 0; nt < 4; ++nt) {
            float s = 0.0f;
#pragma unroll
            for (int mt = 0; mt < 2; ++mt)
#pragma unroll
                for (int v = 0; v < 8; ++v) s += acc[mt][nt][v];
            s += __shfl_xor(s, 16, 32);   // combine lane halves -> 32-agent sum
            csum[nt] = s;
        }
#pragma unroll
        for (int mt = 0; mt < 2; ++mt)
#pragma unroll
            for (int nt = 0; nt < 4; ++nt)
#pragma unroll
                for (int v = 0; v < 8; ++v) {
                    int   m = mt * 16 + half * 8 + v;
                    float h = acc[mt][nt][v];
                    float msg = (csum[nt] - h) * (1.0f / 31.0f);
                    *(unsigned*)(stage + m * SSTR + 2 * (nt * 16 + lr)) =
                        pack2(h, msg);             // 4B aligned packed store
                }
        const bf16_t* wT = s_commT[r];
#pragma unroll
        for (int mt = 0; mt < 2; ++mt)
#pragma unroll
            for (int nt = 0; nt < 4; ++nt) {
                v8f c{};
#pragma unroll
                for (int kt = 0; kt < 4; ++kt) {   // K = 128 (interleaved)
                    const bf16_t* ap = stage + (mt * 16 + lr) * SSTR + kt * 32 + half * 16;
                    v16bf a = load_frag(ap, ap + 8);
                    const bf16_t* bp = wT + (nt * 16 + lr) * (2 * HD) + kt * 32 + half * 16;
                    v16bf b = load_frag(bp, bp + 8);
                    c = __builtin_amdgcn_wmma_f32_16x16x32_bf16(
                            false, a, false, b, (short)0, c, false, false);
                }
                float bias = s_commb[r][nt * 16 + lr];
#pragma unroll
                for (int v = 0; v < 8; ++v) c[v] = fmaxf(c[v] + bias, 0.0f);
                acc[mt][nt] = c;
            }
    }

    // ---- hidden: hid = relu(h @ out_w1 + out_b1), K=64 ----
#pragma unroll
    for (int mt = 0; mt < 2; ++mt)
#pragma unroll
        for (int nt = 0; nt < 4; ++nt)
#pragma unroll
            for (int v = 0; v < 8; ++v) {
                int m = mt * 16 + half * 8 + v;
                stage[m * SSTR + nt * 16 + lr] = f2bf(acc[mt][nt][v]);
            }
#pragma unroll
    for (int mt = 0; mt < 2; ++mt)
#pragma unroll
        for (int nt = 0; nt < 4; ++nt) {
            v8f c{};
#pragma unroll
            for (int kt = 0; kt < 2; ++kt) {
                const bf16_t* ap = stage + (mt * 16 + lr) * SSTR + kt * 32 + half * 16;
                v16bf a = load_frag(ap, ap + 8);
                const bf16_t* bp = s_w1T + (nt * 16 + lr) * HD + kt * 32 + half * 16;
                v16bf b = load_frag(bp, bp + 8);
                c = __builtin_amdgcn_wmma_f32_16x16x32_bf16(
                        false, a, false, b, (short)0, c, false, false);
            }
            float bias = s_b1[nt * 16 + lr];
#pragma unroll
            for (int v = 0; v < 8; ++v) c[v] = fmaxf(c[v] + bias, 0.0f);
            acc[mt][nt] = c;
        }

    // ---- output: q = hid @ out_w2 + out_b2 (N=16), then mask ----
#pragma unroll
    for (int mt = 0; mt < 2; ++mt)
#pragma unroll
        for (int nt = 0; nt < 4; ++nt)
#pragma unroll
            for (int v = 0; v < 8; ++v) {
                int m = mt * 16 + half * 8 + v;
                stage[m * SSTR + nt * 16 + lr] = f2bf(acc[mt][nt][v]);
            }
    v8f q[2];
#pragma unroll
    for (int mt = 0; mt < 2; ++mt) {
        v8f c{};
#pragma unroll
        for (int kt = 0; kt < 2; ++kt) {
            const bf16_t* ap = stage + (mt * 16 + lr) * SSTR + kt * 32 + half * 16;
            v16bf a = load_frag(ap, ap + 8);
            const bf16_t* bp = s_w2T + lr * HD + kt * 32 + half * 16;
            v16bf b = load_frag(bp, bp + 8);
            c = __builtin_amdgcn_wmma_f32_16x16x32_bf16(
                    false, a, false, b, (short)0, c, false, false);
        }
        q[mt] = c;
    }

    float* qb    = q_out + bidx * (NA * AD);
    float  bias2 = s_b2[lr];
#pragma unroll
    for (int mt = 0; mt < 2; ++mt)
#pragma unroll
        for (int v = 0; v < 8; ++v) {
            int   m  = mt * 16 + half * 8 + v;
            float qv = q[mt][v] + bias2;
            int   av = avb[m * AD + lr];
            qb[m * AD + lr] = (av == 0) ? -1e10f : qv;
        }
}

extern "C" void kernel_launch(void* const* d_in, const int* in_sizes, int n_in,
                              void* d_out, int out_size, void* d_ws, size_t ws_size,
                              hipStream_t stream) {
    (void)in_sizes; (void)n_in; (void)out_size; (void)d_ws; (void)ws_size;
    const float* obs    = (const float*)d_in[0];
    const float* enc_w  = (const float*)d_in[1];
    const float* enc_b  = (const float*)d_in[2];
    const float* comm_w = (const float*)d_in[3];
    const float* comm_b = (const float*)d_in[4];
    const float* out_w1 = (const float*)d_in[5];
    const float* out_b1 = (const float*)d_in[6];
    const float* out_w2 = (const float*)d_in[7];
    const float* out_b2 = (const float*)d_in[8];
    const int*   avail  = (const int*)d_in[9];

    dim3 grid(NB / WAVES);        // 2048 blocks, one batch element per wave
    dim3 block(32 * WAVES);       // 8 wave32s per WGP-resident workgroup
    hipLaunchKernelGGL(commnet_fwd, grid, block, 0, stream,
                       obs, enc_w, enc_b, comm_w, comm_b,
                       out_w1, out_b1, out_w2, out_b2, avail, (float*)d_out);
}
